// OuterConcatAttention_80934363726433
// MI455X (gfx1250) — compile-verified, matrix operand order checked by reading, not run on hardware
//
#include <hip/hip_runtime.h>

// ---------------------------------------------------------------------------
// Sizes (fixed by the reference): L=128, N=32, C=128, DIM_MLP=128, OUT=128
// ---------------------------------------------------------------------------
#define LL   128
#define NB   32
#define CC   128
#define ROWS 4096          // L*N flattened rows of x
#define LDH  130           // LDS row stride in halves (padding vs. 64 banks)

typedef __attribute__((ext_vector_type(16))) _Float16 v16h;
typedef __attribute__((ext_vector_type(8)))  float    v8f;

// ---- A-fragment (16x32 f16, row-major source in LDS) per ISA 7.12.2 -------
// lane 0-15  : row = lane,     halves 0-7 -> K 0-7,   halves 8-15 -> K 16-23
// lane 16-31 : row = lane-16,  halves 0-7 -> K 8-15,  halves 8-15 -> K 24-31
__device__ __forceinline__ v16h lds_frag_a(const _Float16* lds, int lane,
                                           int row_base, int k_base) {
  const int r  = row_base + (lane & 15);
  const int kb = k_base + ((lane >> 4) << 3);
  const _Float16* p = lds + r * LDH + kb;
  v16h a;
#pragma unroll
  for (int h = 0; h < 8; ++h) a[h] = p[h];
#pragma unroll
  for (int h = 0; h < 8; ++h) a[8 + h] = p[16 + h];
  return a;
}

// ---- B-fragment (32x16 f16) loaded from a row-major f32 weight [K][ldw] ---
// same K striping as A, lane%16 selects the output column
__device__ __forceinline__ v16h glb_frag_b(const float* __restrict__ W, int lane,
                                           int k_base, int col, int ldw) {
  const int kb = k_base + ((lane >> 4) << 3);
  v16h b;
#pragma unroll
  for (int h = 0; h < 8; ++h) b[h]     = (_Float16)W[(kb + h) * ldw + col];
#pragma unroll
  for (int h = 0; h < 8; ++h) b[8 + h] = (_Float16)W[(kb + 16 + h) * ldw + col];
  return b;
}

__device__ __forceinline__ v8f wmma16(v16h a, v16h b, v8f c) {
  return __builtin_amdgcn_wmma_f32_16x16x32_f16(false, a, false, b,
                                                (short)0, c, false, false);
}

// ---------------------------------------------------------------------------
// Phase 1: P = x@W1a + b1 ; Q = x@W1b ; value = relu(x@Wv1+bv1)@Wv2 + bv2
// grid = 32 blocks x 256 threads, each block owns 128 flat rows of x.
// ---------------------------------------------------------------------------
__global__ __launch_bounds__(256) void oca_prep_kernel(
    const float* __restrict__ x,
    const float* __restrict__ attn_w1, const float* __restrict__ attn_b1,
    const float* __restrict__ value_w1, const float* __restrict__ value_b1,
    const float* __restrict__ value_w2, const float* __restrict__ value_b2,
    float* __restrict__ P, float* __restrict__ Q, float* __restrict__ V) {
  __shared__ _Float16 ldsX[128 * LDH];   // X tile, later reused for hv tile
  const int tid = threadIdx.x;
  const int rb  = blockIdx.x * 128;

#pragma unroll 4
  for (int e = 0; e < 64; ++e) {
    int idx = tid + 256 * e;
    int j = idx >> 7, k = idx & 127;
    ldsX[j * LDH + k] = (_Float16)x[(rb + j) * CC + k];
  }
  __syncthreads();

  const int lane    = tid & 31;
  const int wv      = tid >> 5;
  const int colbase = wv * 16;
  const int col     = colbase + (lane & 15);
  const int roff    = (lane >> 4) << 3;   // +8 rows for upper half-wave

  // ---------------- P = X @ W1a + b1 (W1a = attn_w1 rows 0..127) -----------
  {
    v16h B[4];
#pragma unroll
    for (int ks = 0; ks < 4; ++ks) B[ks] = glb_frag_b(attn_w1, lane, ks * 32, col, 128);
    const float bias = attn_b1[col];
#pragma unroll
    for (int mt = 0; mt < 8; ++mt) {
      v8f acc = {};
#pragma unroll
      for (int ks = 0; ks < 4; ++ks)
        acc = wmma16(lds_frag_a(ldsX, lane, mt * 16, ks * 32), B[ks], acc);
      const int jl = mt * 16 + roff;
#pragma unroll
      for (int r = 0; r < 8; ++r)
        P[(rb + jl + r) * 128 + col] = acc[r] + bias;
    }
  }

  // ---------------- Q = X @ W1b (W1b = attn_w1 rows 128..255) --------------
  {
    v16h B[4];
#pragma unroll
    for (int ks = 0; ks < 4; ++ks) B[ks] = glb_frag_b(attn_w1 + 128 * 128, lane, ks * 32, col, 128);
#pragma unroll
    for (int mt = 0; mt < 8; ++mt) {
      v8f acc = {};
#pragma unroll
      for (int ks = 0; ks < 4; ++ks)
        acc = wmma16(lds_frag_a(ldsX, lane, mt * 16, ks * 32), B[ks], acc);
      const int jl = mt * 16 + roff;
#pragma unroll
      for (int r = 0; r < 8; ++r)
        Q[(rb + jl + r) * 128 + col] = acc[r];
    }
  }

  // ---------------- hv = relu(X @ Wv1 + bv1), held in regs -----------------
  v8f hacc[8];
  {
    v16h B[4];
#pragma unroll
    for (int ks = 0; ks < 4; ++ks) B[ks] = glb_frag_b(value_w1, lane, ks * 32, col, 128);
#pragma unroll
    for (int mt = 0; mt < 8; ++mt) {
      v8f acc = {};
#pragma unroll
      for (int ks = 0; ks < 4; ++ks)
        acc = wmma16(lds_frag_a(ldsX, lane, mt * 16, ks * 32), B[ks], acc);
      hacc[mt] = acc;
    }
  }
  __syncthreads();                       // all reads of X tile done

  const float b1v = value_b1[col];
#pragma unroll
  for (int mt = 0; mt < 8; ++mt) {       // overwrite ldsX with f16 relu(hv)
    const int jl = mt * 16 + roff;
#pragma unroll
    for (int r = 0; r < 8; ++r)
      ldsX[(jl + r) * LDH + col] = (_Float16)fmaxf(hacc[mt][r] + b1v, 0.f);
  }
  __syncthreads();

  // ---------------- value = hv @ Wv2 + bv2 ---------------------------------
  {
    v16h B[4];
#pragma unroll
    for (int ks = 0; ks < 4; ++ks) B[ks] = glb_frag_b(value_w2, lane, ks * 32, col, 128);
    const float bias = value_b2[col];
#pragma unroll
    for (int mt = 0; mt < 8; ++mt) {
      v8f acc = {};
#pragma unroll
      for (int ks = 0; ks < 4; ++ks)
        acc = wmma16(lds_frag_a(ldsX, lane, mt * 16, ks * 32), B[ks], acc);
      const int jl = mt * 16 + roff;
#pragma unroll
      for (int r = 0; r < 8; ++r)
        V[(rb + jl + r) * 128 + col] = acc[r] + bias;
    }
  }
}

// ---------------------------------------------------------------------------
// Phase 2: one block per (i,b).  H[j,k]=relu(P[i,b,k]+Q[j,b,k]) in LDS (f16),
// ATT = H @ W2 via WMMA, fused reduce over j against value[j,b,:].
// grid = (128, 32) x 256 threads.  attn (268 MB in f32) is never materialized.
// ---------------------------------------------------------------------------
__global__ __launch_bounds__(256) void oca_pair_kernel(
    const float* __restrict__ P, const float* __restrict__ Q,
    const float* __restrict__ V,
    const float* __restrict__ attn_w2, const float* __restrict__ attn_b2,
    float* __restrict__ out) {
  __shared__ _Float16 ldsH[128 * LDH];
  __shared__ float sP[128];
  const int i   = blockIdx.x;
  const int b   = blockIdx.y;
  const int tid = threadIdx.x;

  if (tid < 128) sP[tid] = P[(i * NB + b) * 128 + tid];   // bias already folded
  __syncthreads();

#pragma unroll 4
  for (int e = 0; e < 64; ++e) {
    int idx = tid + 256 * e;
    int j = idx >> 7, k = idx & 127;
    float h = sP[k] + Q[(j * NB + b) * 128 + k];
    ldsH[j * LDH + k] = (_Float16)fmaxf(h, 0.f);
  }
  __syncthreads();

  const int lane    = tid & 31;
  const int wv      = tid >> 5;
  const int colbase = wv * 16;
  const int col     = colbase + (lane & 15);
  const int roff    = (lane >> 4) << 3;

  v16h B[4];
#pragma unroll
  for (int ks = 0; ks < 4; ++ks) B[ks] = glb_frag_b(attn_w2, lane, ks * 32, col, 128);
  const float b2 = attn_b2[col];

  float partial = 0.f;
#pragma unroll
  for (int mt = 0; mt < 8; ++mt) {
    v8f acc = {};
#pragma unroll
    for (int ks = 0; ks < 4; ++ks)
      acc = wmma16(lds_frag_a(ldsH, lane, mt * 16, ks * 32), B[ks], acc);
    const int jbase = mt * 16 + roff;
#pragma unroll
    for (int r = 0; r < 8; ++r)   // C/D layout: VGPR r -> row jbase+r, col = lane%16
      partial += (acc[r] + b2) * V[((jbase + r) * NB + b) * 128 + col];
  }

  partial += __shfl_xor(partial, 16, 32);   // combine the two half-wave row groups
  if (lane < 16)
    out[(i * NB + b) * 128 + col] = partial;
}

// ---------------------------------------------------------------------------
extern "C" void kernel_launch(void* const* d_in, const int* in_sizes, int n_in,
                              void* d_out, int out_size, void* d_ws, size_t ws_size,
                              hipStream_t stream) {
  const float* x        = (const float*)d_in[0];
  const float* attn_w1  = (const float*)d_in[1];
  const float* attn_b1  = (const float*)d_in[2];
  const float* attn_w2  = (const float*)d_in[3];
  const float* attn_b2  = (const float*)d_in[4];
  const float* value_w1 = (const float*)d_in[5];
  const float* value_b1 = (const float*)d_in[6];
  const float* value_w2 = (const float*)d_in[7];
  const float* value_b2 = (const float*)d_in[8];

  float* P = (float*)d_ws;                 // 4096*128 f32
  float* Q = P + ROWS * 128;               // 4096*128 f32
  float* V = Q + ROWS * 128;               // 4096*128 f32  (6 MB total)

  oca_prep_kernel<<<dim3(ROWS / 128), dim3(256), 0, stream>>>(
      x, attn_w1, attn_b1, value_w1, value_b1, value_w2, value_b2, P, Q, V);

  oca_pair_kernel<<<dim3(LL, NB), dim3(256), 0, stream>>>(
      P, Q, V, attn_w2, attn_b2, (float*)d_out);
}
